// MultiHeadAttention_71794673319937
// MI455X (gfx1250) — compile-verified, matrix operand order checked by reading, not run on hardware
//
#include <hip/hip_runtime.h>
#include <hip/hip_bf16.h>
#include <math.h>
#include <stdint.h>

// ---------------------------------------------------------------- types
typedef __bf16 v16bf __attribute__((ext_vector_type(16)));
typedef __bf16 bhalf8 __attribute__((ext_vector_type(8)));
typedef float  v8f   __attribute__((ext_vector_type(8)));
typedef int    v4i   __attribute__((ext_vector_type(4)));

union AFrag {
    v16bf  v;      // WMMA operand: elements 2i,2i+1 live in VGPR i
    bhalf8 h[2];   // h[0] = K base+0..7 ; h[1] = K base+16..23 (16B each)
};

static __device__ inline __bf16 f2bf(float f) {
    unsigned u = __builtin_bit_cast(unsigned, f);
    unsigned r = u + 0x7FFFu + ((u >> 16) & 1u);   // round-to-nearest-even
    unsigned short h = (unsigned short)(r >> 16);
    return __builtin_bit_cast(__bf16, h);
}
static __device__ inline float bf2f(__bf16 b) {
    unsigned short u = __builtin_bit_cast(unsigned short, b);
    unsigned x = ((unsigned)u) << 16;
    return __builtin_bit_cast(float, x);
}

// Load a 16x32 bf16 operand fragment from a row-major [row][k] buffer (global or LDS).
// ISA 16-bit operand layout: lanes 0-15 -> K 0..7 & 16..23, lanes 16-31 -> K 8..15 & 24..31.
static __device__ inline void frag_load(AFrag& f, const __bf16* base, int stride, int lane) {
    int r  = lane & 15;
    int kh = (lane >> 4) & 1;
    const __bf16* p = base + (long)r * stride + kh * 8;
    f.h[0] = *(const bhalf8*)(p);
    f.h[1] = *(const bhalf8*)(p + 16);
}

static __device__ inline v8f wmma_bf16(const AFrag& a, const AFrag& b, v8f c) {
    return __builtin_amdgcn_wmma_f32_16x16x32_bf16(
        false, a.v, false, b.v, (short)0, c, false, false);
}

// ---------------------------------------------------------------- async copy to LDS (CDNA5)
#if __has_builtin(__builtin_amdgcn_global_load_async_to_lds_b128)
#define HAS_ASYNC_LDS 1
typedef __attribute__((address_space(1))) v4i g_v4i;   // __device__ (AS1) int4*
typedef __attribute__((address_space(3))) v4i l_v4i;   // __shared__ (AS3) int4*
#endif

static __device__ __forceinline__ void async_copy16(const void* g, void* l) {
#ifdef HAS_ASYNC_LDS
    __builtin_amdgcn_global_load_async_to_lds_b128(
        (g_v4i*)(uintptr_t)g, (l_v4i*)(unsigned)(uintptr_t)l, 0, 0);
#else
    *(bhalf8*)l = *(const bhalf8*)g;   // sync fallback
#endif
}

#if __has_builtin(__builtin_amdgcn_s_wait_asynccnt)
#define WAIT_ASYNC(n) __builtin_amdgcn_s_wait_asynccnt(n)
#else
#define WAIT_ASYNC(n) asm volatile("s_wait_asynccnt " #n ::: "memory")
#endif

#define BB 4
#define HH 16
#define TT 2048
#define CC 1024
#define DD 64
#define SCALE 0.03125f   // 1024^-0.5
#define PAD 40           // LDS row stride (halfwords): r*20 mod 64 -> conflict-free b128 reads

// ---------------------------------------------------------------- 1. fp32 -> bf16
__global__ void k_cvt_bf16(const float* __restrict__ src, __bf16* __restrict__ dst, int n) {
    int i = blockIdx.x * blockDim.x + threadIdx.x;
    if (i < n) dst[i] = f2bf(src[i]);
}

// ---------------------------------------------------------------- 2. W [H,C,D] -> Wt [H,D,C] bf16
__global__ void k_cvt_w_t(const float* __restrict__ w, __bf16* __restrict__ wt, int n) {
    int i = blockIdx.x * blockDim.x + threadIdx.x;
    if (i >= n) return;
    int d = i % DD;
    int t = i / DD;
    int c = t % CC;
    int h = t / CC;
    wt[((long)h * DD + d) * CC + c] = f2bf(w[i]);
}

// ---------------------------------------------------------------- 3. QKV projection (LDS-staged, double-buffered)
__global__ void k_proj(const __bf16* __restrict__ X, const __bf16* __restrict__ Wt,
                       __bf16* __restrict__ out, int transpose_out) {
    __shared__ __bf16 sA[2][128 * PAD];
    __shared__ __bf16 sB[2][64 * PAD];
    int tb = blockIdx.x, h = blockIdx.y, b = blockIdx.z;
    int tid = threadIdx.x, lane = tid & 31, wave = tid >> 5;
    const __bf16* Xb = X + ((long)b * TT + tb * 128) * CC;
    const __bf16* Wh = Wt + (long)h * DD * CC;
    int r0 = tid >> 2, c0 = (tid & 3) * 8;     // 16B chunk per thread

    auto issue = [&](int k, int buf) {
        async_copy16(Xb + (long)r0 * CC + k + c0,        &sA[buf][r0 * PAD + c0]);
        async_copy16(Xb + (long)(r0 + 64) * CC + k + c0, &sA[buf][(r0 + 64) * PAD + c0]);
        async_copy16(Wh + (long)r0 * CC + k + c0,        &sB[buf][r0 * PAD + c0]);
    };

    issue(0, 0);
    v8f acc[4] = {};
    for (int k = 0; k < CC; k += 32) {
        int buf = (k >> 5) & 1;
        if (k + 32 < CC) { issue(k + 32, buf ^ 1); WAIT_ASYNC(3); }
        else             { WAIT_ASYNC(0); }
        __syncthreads();
        AFrag a;
        frag_load(a, &sA[buf][(wave * 16) * PAD], PAD, lane);
        for (int n = 0; n < 4; n++) {
            AFrag bw;
            frag_load(bw, &sB[buf][(n * 16) * PAD], PAD, lane);
            acc[n] = wmma_bf16(a, bw, acc[n]);
        }
        __syncthreads();
    }
    int n_l = lane & 15, mhalf = lane >> 4;
    int t0 = tb * 128 + wave * 16;
    long bh = (long)b * HH + h;
    for (int n = 0; n < 4; n++)
        for (int i = 0; i < 8; i++) {
            int t = t0 + mhalf * 8 + i;
            int d = n * 16 + n_l;
            __bf16 val = f2bf(acc[n][i]);
            if (!transpose_out)
                out[(bh * TT + t) * DD + d] = val;     // [b][h][t][d]
            else
                out[(bh * DD + d) * TT + t] = val;     // [b][h][d][t]
        }
}

// ---------------------------------------------------------------- 4. column softmax stats
// M[s] = max_t sc*q[t].k[s] (t>=s), Z[s] = sum_t exp(sc*qk - M[s])
__global__ void k_stats(const __bf16* __restrict__ Q, const __bf16* __restrict__ K,
                        float* __restrict__ Mx, float* __restrict__ Z) {
    int sb = blockIdx.x, h = blockIdx.y, b = blockIdx.z;
    int lane = threadIdx.x & 31, wave = threadIdx.x >> 5;
    int s0 = sb * 128 + wave * 16;
    long bh = (long)b * HH + h;
    const __bf16* Qb = Q + bh * TT * DD;
    const __bf16* Kb = K + bh * TT * DD;
    AFrag bk0, bk1;   // K rows (N = s), contiguous d; held in registers for whole loop
    frag_load(bk0, Kb + (long)s0 * DD + 0, DD, lane);
    frag_load(bk1, Kb + (long)s0 * DD + 32, DD, lane);
    int n_l = lane & 15, mhalf = lane >> 4;
    int sg = s0 + n_l;
    float mmax = -INFINITY, zsum = 0.f;
    for (int tt = s0 >> 4; tt < TT / 16; tt++) {
        v8f acc = {};
        AFrag a0, a1;
        frag_load(a0, Qb + (long)tt * 16 * DD + 0, DD, lane);
        frag_load(a1, Qb + (long)tt * 16 * DD + 32, DD, lane);
        acc = wmma_bf16(a0, bk0, acc);
        acc = wmma_bf16(a1, bk1, acc);
        float vals[8];
        float tilemax = -INFINITY;
        for (int i = 0; i < 8; i++) {
            int tg = tt * 16 + mhalf * 8 + i;
            float w = SCALE * acc[i];
            vals[i] = (tg >= sg) ? w : -INFINITY;
            tilemax = fmaxf(tilemax, vals[i]);
        }
        if (tilemax != -INFINITY) {
            if (tilemax > mmax) { zsum *= __expf(mmax - tilemax); mmax = tilemax; }
            for (int i = 0; i < 8; i++) zsum += __expf(vals[i] - mmax);
        }
    }
    float mo = __shfl_xor(mmax, 16);
    float zo = __shfl_xor(zsum, 16);
    float M = fmaxf(mmax, mo);
    float Zt = zsum * __expf(mmax - M) + zo * __expf(mo - M);
    if (lane < 16) {
        Mx[bh * TT + sg] = M;
        Z[bh * TT + sg] = Zt;
    }
}

// ---------------------------------------------------------------- 5. V[d][s] /= Z[s]
__global__ void k_scale_v(__bf16* __restrict__ VT, const float* __restrict__ Z, int n) {
    int i = blockIdx.x * blockDim.x + threadIdx.x;
    if (i >= n) return;
    int s = i % TT;
    long r = i / TT;       // r = bh*D + d
    long bh = r / DD;
    float z = Z[bh * TT + s];
    VT[i] = f2bf(bf2f(VT[i]) / z);
}

// ---------------------------------------------------------------- 6. fused attention (LDS-staged K/V, double-buffered)
__global__ void k_attn(const __bf16* __restrict__ Q, const __bf16* __restrict__ K,
                       const __bf16* __restrict__ VT, const float* __restrict__ Mx,
                       __bf16* __restrict__ OH) {
    __shared__ __bf16 sK[2][32 * 72];      // [s][d] rows, stride 72 halfwords
    __shared__ __bf16 sV[2][64 * PAD];     // [d][s] rows, stride 40
    __shared__ __bf16 sP[8][16 * PAD];     // wave-private P tile, [m][k], stride 40
    int tb = blockIdx.x, h = blockIdx.y, b = blockIdx.z;
    int tid = threadIdx.x, lane = tid & 31, wave = tid >> 5;
    int t0 = tb * 128 + wave * 16;
    long bh = (long)b * HH + h;
    const __bf16* Qb = Q + bh * TT * DD;
    const __bf16* Kb = K + bh * TT * DD;
    const __bf16* Vb = VT + bh * DD * TT;   // [d][s]
    const float* Mxb = Mx + bh * TT;
    int n_l = lane & 15, mhalf = lane >> 4;

    AFrag aq0, aq1;   // Q operand held in registers for whole loop
    frag_load(aq0, Qb + (long)t0 * DD + 0, DD, lane);
    frag_load(aq1, Qb + (long)t0 * DD + 32, DD, lane);

    auto issueKV = [&](int s0, int buf) {
        // K chunk: 32 rows(s) x 64(d): 256 x 16B chunks
        async_copy16(Kb + (long)(s0 + (tid >> 3)) * DD + (tid & 7) * 8,
                     &sK[buf][(tid >> 3) * 72 + (tid & 7) * 8]);
        // V chunk: 64 rows(d) x 32(s): 256 x 16B chunks
        async_copy16(Vb + (long)(tid >> 2) * TT + s0 + (tid & 3) * 8,
                     &sV[buf][(tid >> 2) * PAD + (tid & 3) * 8]);
    };

    v8f acc[4] = {};
    int smax = t0 + 16;            // causal limit for this wave
    int send = tb * 128 + 128;     // block-uniform loop bound
    issueKV(0, 0);
    for (int s0 = 0; s0 < send; s0 += 32) {
        int buf = (s0 >> 5) & 1;
        if (s0 + 32 < send) { issueKV(s0 + 32, buf ^ 1); WAIT_ASYNC(2); }
        else                { WAIT_ASYNC(0); }
        __syncthreads();
        if (s0 < smax) {                      // wave-uniform -> EXEC all-ones at WMMA
            for (int half = 0; half < 2; half++) {
                int scol = s0 + half * 16;
                v8f w8 = {};
                if (scol < smax) {
                    AFrag bk0, bk1;
                    frag_load(bk0, &sK[buf][(half * 16) * 72 + 0], 72, lane);
                    frag_load(bk1, &sK[buf][(half * 16) * 72 + 32], 72, lane);
                    w8 = wmma_bf16(aq0, bk0, w8);
                    w8 = wmma_bf16(aq1, bk1, w8);
                }
                int sg = scol + n_l;
                float mcol = (scol < smax) ? Mxb[sg] : 0.f;
                for (int i = 0; i < 8; i++) {
                    int m = mhalf * 8 + i;
                    int tg = t0 + m;
                    float p = 0.f;
                    if (scol < smax && sg <= tg) p = __expf(SCALE * w8[i] - mcol);
                    sP[wave][m * PAD + half * 16 + n_l] = f2bf(p);
                }
            }
            asm volatile("s_wait_dscnt 0" ::: "memory");   // own-wave LDS RAW fence
            AFrag ap;
            frag_load(ap, &sP[wave][0], PAD, lane);        // P as A operand (16x32)
            for (int n = 0; n < 4; n++) {
                AFrag bv;
                frag_load(bv, &sV[buf][(n * 16) * PAD], PAD, lane);
                acc[n] = wmma_bf16(ap, bv, acc[n]);
            }
        }
        __syncthreads();
    }
    for (int n = 0; n < 4; n++)
        for (int i = 0; i < 8; i++) {
            int t = t0 + mhalf * 8 + i;
            int c = h * 64 + n * 16 + n_l;
            OH[((long)b * TT + t) * CC + c] = f2bf(acc[n][i]);
        }
}

// ---------------------------------------------------------------- 7. final projection + bias (LDS-staged)
__global__ void k_final(const __bf16* __restrict__ OH, const __bf16* __restrict__ Wp,
                        const float* __restrict__ bp, float* __restrict__ out) {
    __shared__ __bf16 sA[2][128 * PAD];
    __shared__ __bf16 sB[2][64 * PAD];
    int tb = blockIdx.x, jb = blockIdx.y, b = blockIdx.z;
    int tid = threadIdx.x, lane = tid & 31, wave = tid >> 5;
    int t0 = tb * 128 + wave * 16;
    int j0 = jb * 64;
    const __bf16* Ab = OH + ((long)b * TT + tb * 128) * CC;
    const __bf16* Wb = Wp + (long)j0 * CC;       // rows j, contiguous c  (out @ Wp^T)
    int r0 = tid >> 2, c0 = (tid & 3) * 8;

    auto issue = [&](int k, int buf) {
        async_copy16(Ab + (long)r0 * CC + k + c0,        &sA[buf][r0 * PAD + c0]);
        async_copy16(Ab + (long)(r0 + 64) * CC + k + c0, &sA[buf][(r0 + 64) * PAD + c0]);
        async_copy16(Wb + (long)r0 * CC + k + c0,        &sB[buf][r0 * PAD + c0]);
    };

    int n_l = lane & 15, mhalf = lane >> 4;
    v8f acc[4];
    for (int n = 0; n < 4; n++) {
        float bias = bp[j0 + n * 16 + n_l];
        for (int i = 0; i < 8; i++) acc[n][i] = bias;
    }
    issue(0, 0);
    for (int k = 0; k < CC; k += 32) {
        int buf = (k >> 5) & 1;
        if (k + 32 < CC) { issue(k + 32, buf ^ 1); WAIT_ASYNC(3); }
        else             { WAIT_ASYNC(0); }
        __syncthreads();
        AFrag a;
        frag_load(a, &sA[buf][(wave * 16) * PAD], PAD, lane);
        for (int n = 0; n < 4; n++) {
            AFrag w;
            frag_load(w, &sB[buf][(n * 16) * PAD], PAD, lane);
            acc[n] = wmma_bf16(a, w, acc[n]);
        }
        __syncthreads();
    }
    for (int n = 0; n < 4; n++)
        for (int i = 0; i < 8; i++) {
            int t = t0 + mhalf * 8 + i;
            out[((long)b * TT + t) * CC + j0 + n * 16 + n_l] = acc[n][i];
        }
}

// ---------------------------------------------------------------- launch
extern "C" void kernel_launch(void* const* d_in, const int* in_sizes, int n_in,
                              void* d_out, int out_size, void* d_ws, size_t ws_size,
                              hipStream_t stream) {
    const float* x  = (const float*)d_in[0];
    const float* Wq = (const float*)d_in[1];
    const float* Wk = (const float*)d_in[2];
    const float* Wv = (const float*)d_in[3];
    const float* Wp = (const float*)d_in[4];
    const float* bp = (const float*)d_in[5];
    float* out = (float*)d_out;

    const long nX = (long)BB * TT * CC;          // 8,388,608
    const long nW = (long)HH * CC * DD;          // 1,048,576
    const long nP = (long)BB * HH * TT * DD;     // 8,388,608
    const long nS = (long)BB * HH * TT;          // 131,072

    char* w = (char*)d_ws;
    __bf16* Xbf = (__bf16*)w;  w += nX * 2;
    __bf16* Wqt = (__bf16*)w;  w += nW * 2;
    __bf16* Wkt = (__bf16*)w;  w += nW * 2;
    __bf16* Wvt = (__bf16*)w;  w += nW * 2;
    __bf16* Wpb = (__bf16*)w;  w += (long)CC * CC * 2;
    __bf16* Qb  = (__bf16*)w;  w += nP * 2;
    __bf16* Kb  = (__bf16*)w;  w += nP * 2;
    __bf16* Vt  = (__bf16*)w;  w += nP * 2;
    float*  Mx  = (float*)w;   w += nS * 4;
    float*  Zz  = (float*)w;   w += nS * 4;
    __bf16* OH  = (__bf16*)w;  w += nX * 2;

    dim3 blk(256);
    k_cvt_bf16<<<dim3((nX + 255) / 256), blk, 0, stream>>>(x, Xbf, (int)nX);
    k_cvt_w_t <<<dim3((nW + 255) / 256), blk, 0, stream>>>(Wq, Wqt, (int)nW);
    k_cvt_w_t <<<dim3((nW + 255) / 256), blk, 0, stream>>>(Wk, Wkt, (int)nW);
    k_cvt_w_t <<<dim3((nW + 255) / 256), blk, 0, stream>>>(Wv, Wvt, (int)nW);
    k_cvt_bf16<<<dim3((CC * CC + 255) / 256), blk, 0, stream>>>(Wp, Wpb, CC * CC);

    dim3 gproj(TT / 128, HH, BB);
    k_proj<<<gproj, blk, 0, stream>>>(Xbf, Wqt, Qb, 0);
    k_proj<<<gproj, blk, 0, stream>>>(Xbf, Wkt, Kb, 0);
    k_proj<<<gproj, blk, 0, stream>>>(Xbf, Wvt, Vt, 1);

    k_stats<<<gproj, blk, 0, stream>>>(Qb, Kb, Mx, Zz);
    k_scale_v<<<dim3((nP + 255) / 256), blk, 0, stream>>>(Vt, Zz, (int)nP);
    k_attn<<<gproj, blk, 0, stream>>>(Qb, Kb, Vt, Mx, OH);
    k_final<<<dim3(TT / 128, CC / 64, BB), blk, 0, stream>>>(OH, Wpb, bp, out);
}
